// MoNetLayer_27977416966688
// MI455X (gfx1250) — matrix operand order repaired and checked by reading, not compiled, not run on hardware
//
#include <hip/hip_runtime.h>
#include <hip/hip_bf16.h>
#include <math.h>

typedef __attribute__((ext_vector_type(16))) _Float16 v16h;
typedef __attribute__((ext_vector_type(8)))  _Float16 v8h;
typedef __attribute__((ext_vector_type(8)))  float    v8f;
typedef __attribute__((ext_vector_type(4)))  float    v4f;

namespace {
constexpr int Bn = 32, Nn = 256, Kn = 25, Fn = 64;
}

// One block per (batch b, 16-row i-tile). 8 waves, each wave owns gaussians
// k = wave, wave+8, ... Aggregation and FC both run on v_wmma_f32_16x16x32_f16.
// coord tile and x[b] are staged in LDS; fc_W rows are prefetched per k.
__global__ __launch_bounds__(256) void monet_fused_kernel(
    const float* __restrict__ x,           // [B,N,F]
    const float* __restrict__ coord,       // [B,N,N,2] (rho,theta; NaN = no edge)
    const float* __restrict__ mask,        // [B,N]
    const float* __restrict__ coords_mu,   // [2,K] (row 0 = mu_rho; mu_theta unused per upstream bug)
    const float* __restrict__ sigma_rho,   // [K]
    const float* __restrict__ sigma_theta, // [K]
    const float* __restrict__ fc_W,        // [F_OUT, K*F]
    const float* __restrict__ fc_b,        // [F_OUT]
    float* __restrict__ out)               // [B,N,F_OUT]
{
    __shared__ _Float16 x_t[Fn * Nn];        // x[b] transposed -> [f][j], f16 (32 KB)
    __shared__ float2   coord_s[16 * Nn];    // coord tile [i_local][j] (32 KB)
    __shared__ _Float16 agg16[8 * 16 * Fn];  // per-wave 16x64 agg tile, f16 (16 KB)
    __shared__ float    out_acc[16 * Fn];    // block-level k-reduction (4 KB)

    const int tid  = threadIdx.x;
    const int wave = tid >> 5;
    const int lane = tid & 31;
    const int grp  = lane >> 4;   // lane group 0/1 (CDNA5 wmma half-wave split)
    const int lrow = lane & 15;
    const int b    = blockIdx.x >> 4;
    const int i0   = (blockIdx.x & 15) * 16;

    // zero the block accumulator
    for (int idx = tid; idx < 16 * Fn; idx += 256) out_acc[idx] = 0.0f;

    // stage x[b] into LDS transposed + converted to f16 so wmma-B operands
    // are contiguous 8-half runs (ds_load_b128)
    const float* xb = x + (size_t)b * Nn * Fn;
    for (int idx = tid; idx < Nn * Fn; idx += 256) {
        const int j = idx >> 6, f = idx & 63;
        x_t[f * Nn + j] = (_Float16)xb[idx];
    }
    // stage the 16x256 coord tile (raw float2, NaN pattern preserved) into LDS:
    // coalesced 8B global loads once per block; scattered re-reads then hit LDS
    const float2* cg = (const float2*)coord + ((size_t)b * Nn + i0) * Nn;
    for (int idx = tid; idx < 16 * Nn; idx += 256) coord_s[idx] = cg[idx];
    __syncthreads();

    const float TWO_PI = 6.28318530717958647692f;
    const float2* crow = &coord_s[lrow * Nn];
    _Float16* aw = &agg16[wave * 16 * Fn];

    v8f d2[4] = {};  // FC output tile 16 x 64 f32, accumulated over this wave's k's

    for (int k = wave; k < Kn; k += 8) {
        // speculative prefetch of this k's fc_W slices (hidden behind the j-loop)
        #pragma unroll
        for (int t2 = 0; t2 < 4; ++t2)
            __builtin_prefetch(fc_W + (size_t)(t2 * 16 + lrow) * (Kn * Fn) + k * Fn
                                    + grp * 32, 0, 3);

        const float mu = coords_mu[k];          // mu_rho (also used for theta, per reference)
        const float sr = sigma_rho[k];
        const float st = sigma_theta[k];
        const float ar = -0.5f / (1e-14f + sr * sr);
        const float at = -0.5f / (1e-14f + st * st);

        v8f acc[4] = {};  // agg tile 16 x 64 f32
        for (int jc = 0; jc < Nn; jc += 32) {
            // ---- build weight A-tile (16 x 32 f16) in CDNA5 A layout ----
            // lane = row i; half h -> VGPR v=h>>1: K = 2v+(h&1) + grp*8 + (v>=4 ? 8 : 0)
            v16h A;
            #pragma unroll
            for (int h = 0; h < 16; ++h) {
                const int v  = h >> 1;
                const int jl = 2 * v + (h & 1) + grp * 8 + ((v >= 4) ? 8 : 0);
                const float2 c2 = crow[jc + jl];
                const bool edge = !(c2.x != c2.x);          // !isnan(rho)
                const float r   = edge ? c2.x : 0.0f;
                const float th  = edge ? c2.y : 0.0f;
                const float dr  = r - mu;
                const float wr  = __expf(ar * dr * dr);     // v_exp_f32
                const float dd  = fabsf(th - mu);
                const float an  = fminf(dd, fabsf(TWO_PI - dd));
                const float wt  = __expf(at * an * an);
                A[h] = (_Float16)(edge ? wr * wt : 0.0f);
            }
            // ---- 4 column tiles of x: B is 32 x 16, K contiguous in LDS ----
            #pragma unroll
            for (int t = 0; t < 4; ++t) {
                union { v16h v; v8h h2[2]; } Bm;
                const v8h* p = (const v8h*)&x_t[(t * 16 + lrow) * Nn + jc + grp * 8];
                Bm.h2[0] = p[0];
                Bm.h2[1] = p[2];
                acc[t] = __builtin_amdgcn_wmma_f32_16x16x32_f16(
                    false, A, false, Bm.v, (short)0, acc[t], false, false);
            }
        }

        // ---- spill agg tile (D layout: lane=col, vgpr=row) to LDS as f16 ----
        #pragma unroll
        for (int t = 0; t < 4; ++t)
            #pragma unroll
            for (int r = 0; r < 8; ++r)
                aw[(r + grp * 8) * Fn + t * 16 + lrow] = (_Float16)acc[t][r];
        __asm__ volatile("s_wait_dscnt 0x0" ::: "memory");

        // ---- FC: out_tile += agg_k (16x64) x W_k^T (64x64), K split in two 32-chunks ----
        #pragma unroll
        for (int c = 0; c < 2; ++c) {
            union { v16h v; v8h h2[2]; } A2;   // re-read agg in A layout: two b128s
            const _Float16* ar16 = &aw[lrow * Fn + c * 32 + grp * 16];
            A2.h2[0] = *(const v8h*)ar16;
            A2.h2[1] = *(const v8h*)(ar16 + 8);
            #pragma unroll
            for (int t2 = 0; t2 < 4; ++t2) {
                // B[f, fo]: lane = fo = t2*16+lrow; 16 contiguous f values from fc_W row
                const float* wrow = fc_W + (size_t)(t2 * 16 + lrow) * (Kn * Fn)
                                 + k * Fn + c * 32 + grp * 16;
                const v4f w0 = ((const v4f*)wrow)[0];
                const v4f w1 = ((const v4f*)wrow)[1];
                const v4f w2 = ((const v4f*)wrow)[2];
                const v4f w3 = ((const v4f*)wrow)[3];
                v16h Bw;
                #pragma unroll
                for (int q = 0; q < 4; ++q) {
                    Bw[q]      = (_Float16)w0[q];
                    Bw[4 + q]  = (_Float16)w1[q];
                    Bw[8 + q]  = (_Float16)w2[q];
                    Bw[12 + q] = (_Float16)w3[q];
                }
                d2[t2] = __builtin_amdgcn_wmma_f32_16x16x32_f16(
                    false, A2.v, false, Bw, (short)0, d2[t2], false, false);
            }
        }
    }

    // ---- reduce the 8 waves' k-partials into the block accumulator ----
    #pragma unroll
    for (int t2 = 0; t2 < 4; ++t2)
        #pragma unroll
        for (int r = 0; r < 8; ++r)
            atomicAdd(&out_acc[(r + grp * 8) * Fn + t2 * 16 + lrow], d2[t2][r]);
    __syncthreads();

    // ---- bias + mask, write out ----
    for (int idx = tid; idx < 16 * Fn; idx += 256) {
        const int il = idx >> 6, fo = idx & 63;
        const int i  = i0 + il;
        out[((size_t)b * Nn + i) * Fn + fo] =
            (out_acc[idx] + fc_b[fo]) * mask[b * Nn + i];
    }
}

extern "C" void kernel_launch(void* const* d_in, const int* in_sizes, int n_in,
                              void* d_out, int out_size, void* d_ws, size_t ws_size,
                              hipStream_t stream) {
    const float* x            = (const float*)d_in[0];
    const float* coord        = (const float*)d_in[1];
    const float* mask         = (const float*)d_in[2];
    const float* coords_mu    = (const float*)d_in[3];
    const float* sigma_rho    = (const float*)d_in[4];
    const float* sigma_theta  = (const float*)d_in[5];
    const float* fc_W         = (const float*)d_in[6];
    const float* fc_b         = (const float*)d_in[7];
    float* out = (float*)d_out;

    const dim3 grid(Bn * (Nn / 16));  // 512 blocks: (b, i-tile)
    monet_fused_kernel<<<grid, 256, 0, stream>>>(
        x, coord, mask, coords_mu, sigma_rho, sigma_theta, fc_W, fc_b, out);
}
